// Decoder_77403900609161
// MI455X (gfx1250) — matrix operand order repaired and checked by reading, not compiled, not run on hardware
//
#include <hip/hip_runtime.h>
#include <hip/hip_bf16.h>
#include <math.h>

#define Bb 256
#define Ll 128
#define Ee 512
#define Hh 512

typedef __attribute__((ext_vector_type(16))) __bf16 v16bf;
typedef __attribute__((ext_vector_type(8)))  float  v8f;
typedef __attribute__((ext_vector_type(4)))  unsigned int u32x4;

__device__ __forceinline__ __bf16 f2bf(float f) {
    unsigned u = __builtin_bit_cast(unsigned, f);
    u += 0x7FFFu + ((u >> 16) & 1u);           // round-to-nearest-even
    unsigned short h = (unsigned short)(u >> 16);
    return __builtin_bit_cast(__bf16, h);
}
__device__ __forceinline__ float sigf(float x) { return 1.f / (1.f + __expf(-x)); }

// ---- WMMA fragment loaders (CDNA5 16-bit layouts, CDNA5 ISA §7.12.2) ----
// A (16x32, MxK) row-major: lane&15 = row; groups of 8 consecutive K at
//   k0 + 8*(lane>>4) and k0 + 16 + 8*(lane>>4)
__device__ __forceinline__ v16bf load_a_frag(const __bf16* A, int lda, int m0, int k0, int lane) {
    int m  = m0 + (lane & 15);
    int kh = (lane >> 4) << 3;
    const __bf16* p = A + (size_t)m * lda + k0 + kh;
    union { u32x4 q[2]; v16bf v; } f;
    f.q[0] = *(const u32x4*)p;
    f.q[1] = *(const u32x4*)(p + 16);
    return f.v;
}
// B (32x16, KxN) supplied as row-major [N,K] (i.e., W[n,k]): lane&15 = column n,
// groups of 8 consecutive K at k0 + 8*(lane>>4) and k0 + 16 + 8*(lane>>4)
__device__ __forceinline__ v16bf load_b_frag(const __bf16* Brow, int ldb, int n0, int k0, int lane) {
    int n  = n0 + (lane & 15);
    int kl = k0 + ((lane >> 4) << 3);
    const __bf16* p = Brow + (size_t)n * ldb + kl;
    union { u32x4 q[2]; v16bf v; } f;
    f.q[0] = *(const u32x4*)p;
    f.q[1] = *(const u32x4*)(p + 16);
    return f.v;
}
__device__ __forceinline__ v8f wmma_bf16(v16bf a, v16bf b, v8f c) {
    return __builtin_amdgcn_wmma_f32_16x16x32_bf16(false, a, false, b, (short)0, c, false, false);
}

// ---- Generic dual-source GEMM: C[M,N] = A1@B1^T + A2@B2^T (+bias, opt tanh, opt bf16 copy)
// One wave computes a 32(M) x 64(N) strip: 8 accumulators, A reuse x4, B reuse x2.
__global__ __launch_bounds__(256) void gemm_dual_kernel(
    const __bf16* __restrict__ A1, const __bf16* __restrict__ B1, int ldb1, int K1,
    const __bf16* __restrict__ A2, const __bf16* __restrict__ B2, int ldb2, int K2,
    const float* __restrict__ bias, float* __restrict__ C, __bf16* __restrict__ Cbf,
    int M, int N, int epi_tanh)
{
    int lane = threadIdx.x & 31;
    int wid  = blockIdx.x * 8 + (threadIdx.x >> 5);
    int nStrips = N >> 6;
    int m0 = (wid / nStrips) << 5;     // 32-row M tile
    int n0 = (wid % nStrips) << 6;     // 64-col N strip

    v8f zero = {0.f,0.f,0.f,0.f,0.f,0.f,0.f,0.f};
    v8f acc[8] = {zero, zero, zero, zero, zero, zero, zero, zero};

    for (int k0 = 0; k0 < K1; k0 += 32) {
        if (k0 + 64 < K1) {            // prefetch next-next A chunk (global_prefetch_b8)
            __builtin_prefetch(A1 + (size_t)(m0 + (lane & 15)) * K1 + k0 + 64, 0, 1);
            __builtin_prefetch(A1 + (size_t)(m0 + 16 + (lane & 15)) * K1 + k0 + 64, 0, 1);
        }
        v16bf a0 = load_a_frag(A1, K1, m0,      k0, lane);
        v16bf a1 = load_a_frag(A1, K1, m0 + 16, k0, lane);
#pragma unroll
        for (int j = 0; j < 4; ++j) {
            v16bf b = load_b_frag(B1, ldb1, n0 + (j << 4), k0, lane);
            acc[j]     = wmma_bf16(a0, b, acc[j]);
            acc[4 + j] = wmma_bf16(a1, b, acc[4 + j]);
        }
    }
    if (A2) {
        for (int k0 = 0; k0 < K2; k0 += 32) {
            if (k0 + 64 < K2) {
                __builtin_prefetch(A2 + (size_t)(m0 + (lane & 15)) * K2 + k0 + 64, 0, 1);
                __builtin_prefetch(A2 + (size_t)(m0 + 16 + (lane & 15)) * K2 + k0 + 64, 0, 1);
            }
            v16bf a0 = load_a_frag(A2, K2, m0,      k0, lane);
            v16bf a1 = load_a_frag(A2, K2, m0 + 16, k0, lane);
#pragma unroll
            for (int j = 0; j < 4; ++j) {
                v16bf b = load_b_frag(B2, ldb2, n0 + (j << 4), k0, lane);
                acc[j]     = wmma_bf16(a0, b, acc[j]);
                acc[4 + j] = wmma_bf16(a1, b, acc[4 + j]);
            }
        }
    }
    int rbase = (lane >> 4) << 3;      // C layout: VGPR r -> M = r + 8*(lane>=16)
    int cbase = lane & 15;             //           N = lane & 15
#pragma unroll
    for (int half = 0; half < 2; ++half) {
#pragma unroll
        for (int j = 0; j < 4; ++j) {
            int n = n0 + (j << 4) + cbase;
            float bia = bias ? bias[n] : 0.f;
#pragma unroll
            for (int r = 0; r < 8; ++r) {
                int m = m0 + (half << 4) + rbase + r;
                float v = acc[half * 4 + j][r] + bia;
                if (epi_tanh) v = tanhf(v);
                C[(size_t)m * N + n] = v;
                if (Cbf) Cbf[(size_t)m * N + n] = f2bf(v);
            }
        }
    }
}

// ---- Batched ctx projection: ctx_proj[b,k,l] = sum_h W_ctx[k,h]*context[b,l,h] + b_ctx[k]
// One wave: 32(k) x 64(l) strip; 32 waves per batch row.
__global__ __launch_bounds__(256) void gemm_ctx_kernel(
    const __bf16* __restrict__ Wctx, const __bf16* __restrict__ ctxbf,
    const float* __restrict__ bctx, float* __restrict__ ctx_proj)
{
    int lane = threadIdx.x & 31;
    int wid  = blockIdx.x * 8 + (threadIdx.x >> 5);
    int b  = wid >> 5;                 // 32 waves per batch row
    int w  = wid & 31;
    int m0 = (w >> 1) << 5;            // 16 M-tiles of 32 (k dim = 512)
    int n0 = (w & 1) << 6;             // 2 N-strips of 64 (l dim = 128)
    const __bf16* Brow = ctxbf + (size_t)b * Ll * Hh;   // row n=l contiguous in h

    v8f zero = {0.f,0.f,0.f,0.f,0.f,0.f,0.f,0.f};
    v8f acc[8] = {zero, zero, zero, zero, zero, zero, zero, zero};
    for (int k0 = 0; k0 < Hh; k0 += 32) {
        if (k0 + 64 < Hh) {
            __builtin_prefetch(Wctx + (size_t)(m0 + (lane & 15)) * Hh + k0 + 64, 0, 1);
            __builtin_prefetch(Wctx + (size_t)(m0 + 16 + (lane & 15)) * Hh + k0 + 64, 0, 1);
        }
        v16bf a0 = load_a_frag(Wctx, Hh, m0,      k0, lane);
        v16bf a1 = load_a_frag(Wctx, Hh, m0 + 16, k0, lane);
#pragma unroll
        for (int j = 0; j < 4; ++j) {
            v16bf bf = load_b_frag(Brow, Hh, n0 + (j << 4), k0, lane);
            acc[j]     = wmma_bf16(a0, bf, acc[j]);
            acc[4 + j] = wmma_bf16(a1, bf, acc[4 + j]);
        }
    }
    int rbase = (lane >> 4) << 3;
    int cbase = lane & 15;
#pragma unroll
    for (int half = 0; half < 2; ++half) {
#pragma unroll
        for (int j = 0; j < 4; ++j) {
            int n = n0 + (j << 4) + cbase;
#pragma unroll
            for (int r = 0; r < 8; ++r) {
                int m = m0 + (half << 4) + rbase + r;
                ctx_proj[((size_t)b * Hh + m) * Ll + n] = acc[half * 4 + j][r] + bctx[m];
            }
        }
    }
}

__global__ void cast_kernel(const float* __restrict__ src, __bf16* __restrict__ dst, int n) {
    int i = blockIdx.x * 256 + threadIdx.x;
    if (i < n) dst[i] = f2bf(src[i]);
}

__global__ void init_kernel(const float* __restrict__ hidden_c, const float* __restrict__ b_ih,
                            const float* __restrict__ b_hh, const float* __restrict__ add_mask,
                            float* __restrict__ c, float* __restrict__ mask,
                            float* __restrict__ gbias, float* __restrict__ am_sum)
{
    int i = blockIdx.x * 256 + threadIdx.x;
    if (i < Bb * Hh) c[i] = hidden_c[i];
    if (i < Bb * Ll) mask[i] = 1.f;
    if (i < 4 * Hh)  gbias[i] = b_ih[i] + b_hh[i];
    if (i < Bb) {
        float s = 0.f;
        for (int l = 0; l < Ll; ++l) s += add_mask[i * Ll + l];
        am_sum[i] = s;
    }
}

__global__ __launch_bounds__(256) void lstm_kernel(const float* __restrict__ gates,
                                                   float* __restrict__ c, __bf16* __restrict__ ht_bf)
{
    int idx = blockIdx.x * 256 + threadIdx.x;        // B*H
    int b = idx >> 9, j = idx & 511;
    const float* g = gates + (size_t)b * 2048;
    float ig = sigf(g[j]);
    float fg = sigf(g[512 + j]);
    float gg = tanhf(g[1024 + j]);
    float og = sigf(g[1536 + j]);
    float cn = fg * c[idx] + ig * gg;
    c[idx] = cn;
    ht_bf[idx] = f2bf(og * tanhf(cn));
}

__global__ __launch_bounds__(128) void att_scores_kernel(
    const float* __restrict__ q, const float* __restrict__ ctx_proj,
    const float* __restrict__ V, const float* __restrict__ mask, float* __restrict__ att)
{
    int b = blockIdx.x, l = threadIdx.x;
    const float* qb = q + (size_t)b * Hh;
    const float* cb = ctx_proj + (size_t)b * Hh * Ll + l;
    float s = 0.f;
#pragma unroll 4
    for (int k = 0; k < Hh; ++k)
        s += V[k] * tanhf(qb[k] + cb[(size_t)k * Ll]);
    att[b * Ll + l] = (mask[b * Ll + l] == 0.f) ? -__builtin_huge_valf() : s;
}

__global__ __launch_bounds__(128) void softmax_pointer_kernel(
    const float* __restrict__ att, const float* __restrict__ add_mask,
    const float* __restrict__ am_sum, const float* __restrict__ embedded,
    float* __restrict__ mask, float* __restrict__ alpha_out,
    __bf16* __restrict__ dec_bf, float* __restrict__ out, int t)
{
    __shared__ float sred[Ll];
    __shared__ int   sidx[Ll];
    __shared__ float smsk[Ll];
    int b = blockIdx.x, l = threadIdx.x;
    float a = att[b * Ll + l];

    sred[l] = a; __syncthreads();
    for (int s = 64; s > 0; s >>= 1) { if (l < s) sred[l] = fmaxf(sred[l], sred[l + s]); __syncthreads(); }
    float mx = sred[0]; __syncthreads();

    float e = (a == -__builtin_huge_valf()) ? 0.f : __expf(a - mx);
    sred[l] = e; __syncthreads();
    for (int s = 64; s > 0; s >>= 1) { if (l < s) sred[l] += sred[l + s]; __syncthreads(); }
    float alpha = e / sred[0]; __syncthreads();
    alpha_out[b * Ll + l] = alpha;

    float normed = (alpha + 1e-8f) * add_mask[b * Ll + l];
    sred[l] = normed; __syncthreads();
    for (int s = 64; s > 0; s >>= 1) { if (l < s) sred[l] += sred[l + s]; __syncthreads(); }
    normed /= sred[0]; __syncthreads();

    float mold = mask[b * Ll + l];
    float nm = mold * ((am_sum[b] >= (float)t) ? 1.f : 0.f);
    float masked = normed * nm;
    smsk[l] = masked;

    sred[l] = masked; sidx[l] = l; __syncthreads();
    for (int s = 64; s > 0; s >>= 1) {
        if (l < s) {
            float vo = sred[l + s]; int io = sidx[l + s];
            if (vo > sred[l] || (vo == sred[l] && io < sidx[l])) { sred[l] = vo; sidx[l] = io; }
        }
        __syncthreads();
    }
    int action = sidx[0]; __syncthreads();

    sred[l] = masked + 1e-8f; __syncthreads();
    for (int s = 64; s > 0; s >>= 1) { if (l < s) sred[l] += sred[l + s]; __syncthreads(); }
    if (l == 0) {
        float logp = __logf(smsk[action] + 1e-8f) - __logf(sred[0]);
        out[(size_t)b * Ll + t] = logp;                                // outputs
        out[(size_t)Bb * Ll + (size_t)b * Ll + t] = (float)action;     // pointers
    }
    mask[b * Ll + l] = (l == action) ? 0.f : mold;

    const float* src = embedded + ((size_t)b * Ll + action) * Ee;      // gather next dec input
    for (int e2 = l; e2 < Ee; e2 += Ll) dec_bf[(size_t)b * Ee + e2] = f2bf(src[e2]);
}

__global__ __launch_bounds__(256) void hidden_gemv_kernel(
    const float* __restrict__ ctx_proj, const float* __restrict__ alpha, __bf16* __restrict__ hid_bf)
{
    int idx = blockIdx.x * 256 + threadIdx.x;        // B*H
    int b = idx >> 9, k = idx & 511;
    const float* row = ctx_proj + ((size_t)b * Hh + k) * Ll;
    const float* al  = alpha + b * Ll;
    float s = 0.f;
#pragma unroll 4
    for (int l = 0; l < Ll; ++l) s += row[l] * al[l];
    hid_bf[idx] = f2bf(s);
}

__global__ void final_copy_kernel(const float* __restrict__ h, const float* __restrict__ c,
                                  float* __restrict__ out)
{
    int i = blockIdx.x * 256 + threadIdx.x;
    if (i < Bb * Hh) {
        out[2 * Bb * Ll + i] = h[i];
        out[2 * Bb * Ll + Bb * Hh + i] = c[i];
    }
}

extern "C" void kernel_launch(void* const* d_in, const int* in_sizes, int n_in,
                              void* d_out, int out_size, void* d_ws, size_t ws_size,
                              hipStream_t stream)
{
    const float* embedded = (const float*)d_in[0];
    const float* dec_in   = (const float*)d_in[1];
    const float* hidden_h = (const float*)d_in[2];
    const float* hidden_c = (const float*)d_in[3];
    const float* context  = (const float*)d_in[4];
    const float* add_mask = (const float*)d_in[5];
    const float* W_ih  = (const float*)d_in[6];
    const float* W_hh  = (const float*)d_in[7];
    const float* b_ih  = (const float*)d_in[8];
    const float* b_hh  = (const float*)d_in[9];
    const float* W_out = (const float*)d_in[10];
    const float* b_out = (const float*)d_in[11];
    const float* W_in  = (const float*)d_in[12];
    const float* b_in  = (const float*)d_in[13];
    const float* W_ctx = (const float*)d_in[14];
    const float* b_ctx = (const float*)d_in[15];
    const float* V     = (const float*)d_in[16];
    float* out = (float*)d_out;

    char* base = (char*)d_ws;
    size_t off = 0;
    auto alloc = [&](size_t bytes) { void* r = base + off; off += (bytes + 255) & ~(size_t)255; return r; };

    float*  ctx_proj = (float*) alloc((size_t)Bb * Hh * Ll * 4);   // 67 MB (L2-resident)
    __bf16* ctx_bf   = (__bf16*)alloc((size_t)Bb * Ll * Hh * 2);   // 33 MB
    __bf16* Wih_bf   = (__bf16*)alloc((size_t)4 * Hh * Ee * 2);
    __bf16* Whh_bf   = (__bf16*)alloc((size_t)4 * Hh * Hh * 2);
    __bf16* Win_bf   = (__bf16*)alloc((size_t)Hh * Hh * 2);
    __bf16* Wout_bf  = (__bf16*)alloc((size_t)Hh * 2 * Hh * 2);
    __bf16* Wctx_bf  = (__bf16*)alloc((size_t)Hh * Hh * 2);
    float*  gates    = (float*) alloc((size_t)Bb * 4 * Hh * 4);
    float*  cbuf     = (float*) alloc((size_t)Bb * Hh * 4);
    float*  hbuf     = (float*) alloc((size_t)Bb * Hh * 4);
    __bf16* h_bf     = (__bf16*)alloc((size_t)Bb * Hh * 2);
    __bf16* ht_bf    = (__bf16*)alloc((size_t)Bb * Hh * 2);
    __bf16* hid_bf   = (__bf16*)alloc((size_t)Bb * Hh * 2);
    __bf16* dec_bf   = (__bf16*)alloc((size_t)Bb * Ee * 2);
    float*  qbuf     = (float*) alloc((size_t)Bb * Hh * 4);
    float*  attb     = (float*) alloc((size_t)Bb * Ll * 4);
    float*  alphab   = (float*) alloc((size_t)Bb * Ll * 4);
    float*  maskb    = (float*) alloc((size_t)Bb * Ll * 4);
    float*  am_sum   = (float*) alloc((size_t)Bb * 4);
    float*  gbias    = (float*) alloc((size_t)4 * Hh * 4);
    (void)ws_size; (void)in_sizes; (void)n_in; (void)out_size;

    auto cast = [&](const float* s, __bf16* d, int n) {
        cast_kernel<<<(n + 255) / 256, 256, 0, stream>>>(s, d, n);
    };
    cast(W_ih,  Wih_bf,  4 * Hh * Ee);
    cast(W_hh,  Whh_bf,  4 * Hh * Hh);
    cast(W_in,  Win_bf,  Hh * Hh);
    cast(W_out, Wout_bf, Hh * 2 * Hh);
    cast(W_ctx, Wctx_bf, Hh * Hh);
    cast(context, ctx_bf, Bb * Ll * Hh);
    cast(dec_in,  dec_bf, Bb * Ee);
    cast(hidden_h, h_bf,  Bb * Hh);
    init_kernel<<<(Bb * Hh + 255) / 256, 256, 0, stream>>>(hidden_c, b_ih, b_hh, add_mask,
                                                           cbuf, maskb, gbias, am_sum);
    // ctx_proj: B batched 512x128x512 GEMMs, 32 waves per batch row -> 1024 blocks
    gemm_ctx_kernel<<<Bb * 4, 256, 0, stream>>>(Wctx_bf, ctx_bf, b_ctx, ctx_proj);

    for (int t = 0; t < Ll; ++t) {
        // gates = dec @ Wih^T + h @ Whh^T + (b_ih + b_hh)   [256 x 2048, K=512+512]
        gemm_dual_kernel<<<(Bb / 32) * (4 * Hh / 64) / 8, 256, 0, stream>>>(
            dec_bf, Wih_bf, Ee, Ee, h_bf, Whh_bf, Hh, Hh,
            gbias, gates, nullptr, Bb, 4 * Hh, 0);
        lstm_kernel<<<Bb * Hh / 256, 256, 0, stream>>>(gates, cbuf, ht_bf);
        // q = h_t @ Win^T + b_in   [256 x 512, K=512]
        gemm_dual_kernel<<<(Bb / 32) * (Hh / 64) / 8, 256, 0, stream>>>(
            ht_bf, Win_bf, Hh, Hh, nullptr, nullptr, 0, 0,
            b_in, qbuf, nullptr, Bb, Hh, 0);
        att_scores_kernel<<<Bb, Ll, 0, stream>>>(qbuf, ctx_proj, V, maskb, attb);
        softmax_pointer_kernel<<<Bb, Ll, 0, stream>>>(attb, add_mask, am_sum, embedded,
                                                      maskb, alphab, dec_bf, out, t);
        hidden_gemv_kernel<<<Bb * Hh / 256, 256, 0, stream>>>(ctx_proj, alphab, hid_bf);
        // h_new = tanh([hidden_t, h_t] @ Wout^T + b_out)   [256 x 512, K=512+512]
        gemm_dual_kernel<<<(Bb / 32) * (Hh / 64) / 8, 256, 0, stream>>>(
            hid_bf, Wout_bf, 2 * Hh, Hh, ht_bf, Wout_bf + Hh, 2 * Hh, Hh,
            b_out, hbuf, h_bf, Bb, Hh, 1);
    }
    final_copy_kernel<<<Bb * Hh / 256, 256, 0, stream>>>(hbuf, cbuf, out);
}
